// VectorQuantizer_19275813225065
// MI455X (gfx1250) — matrix operand order repaired and checked by reading, not compiled, not run on hardware
//
#include <hip/hip_runtime.h>

typedef __attribute__((ext_vector_type(16))) __bf16 v16bf;
typedef __attribute__((ext_vector_type(8)))  float  v8f;
typedef __attribute__((ext_vector_type(4)))  unsigned int v4ui;
typedef __attribute__((ext_vector_type(8)))  unsigned int v8ui;

#define N_EMB 1024
#define CDIM 64
#define SEQ 4096
#define BS 32
#define ROWS_PER_BLOCK 128
#define THREADS 256
#define ZP 72                      // padded f32 stride for z tile rows in LDS
#define CHUNK_CODES 64
#define NCHUNK (N_EMB / CHUNK_CODES)
#define CODE_GUINTS 64             // packed bf16 per code in global: 32 hi + 32 lo dwords
#define CHUNK_GUINTS (CHUNK_CODES * CODE_GUINTS)   // 4096 dwords = 16 KiB per chunk
#define EPW 68                     // padded dword stride per code row in LDS (272 B)
#define CHUNK_LDS_BYTES (CHUNK_CODES * EPW * 4)    // 17408 B
#define SMEM_ESQ_OFF 36864
#define SMEM_IDX_OFF 40960
#define SMEM_BYTES 41472

// ---------- helpers ----------
__device__ __forceinline__ unsigned f2bf(float x) {        // RNE f32 -> bf16 bits
  unsigned u = __float_as_uint(x);
  return (u + 0x7FFFu + ((u >> 16) & 1u)) >> 16;
}
__device__ __forceinline__ float bf2f(unsigned b) { return __uint_as_float(b << 16); }

struct U32x8 { uint4 a, b; };
__device__ __forceinline__ v16bf ldbf16x16(const unsigned* p) {  // 16 bf16 from LDS
  U32x8 t;
  t.a = *(const uint4*)(p);
  t.b = *(const uint4*)(p + 4);
  return __builtin_bit_cast(v16bf, t);
}

// TDM: 1-D copy of 2048 x 8B elements (16 KiB) global->LDS, inserting 16 B of
// LDS padding every 256 B (pad_interval=5 -> 256 B, pad_amount=3 -> 4 dwords),
// i.e. each 256 B code row lands on a 272 B stride for bank-conflict-free reads.
__device__ __forceinline__ void tdm_load_chunk(const void* gsrc, unsigned lds_off) {
  unsigned long long ga = (unsigned long long)gsrc;
  v4ui g0;
  g0[0] = 1u;                                            // count=1, no gather
  g0[1] = lds_off;                                       // lds_addr
  g0[2] = (unsigned)ga;                                  // global_addr[31:0]
  g0[3] = ((unsigned)(ga >> 32) & 0x01FFFFFFu) | (2u << 30);  // addr[56:32] | type=2
  v8ui g1;
  g1[0] = (3u << 16) | (1u << 20) | (5u << 22) | (3u << 25);  // data_size=8B, pad_en, int=256B, amt=4dw
  g1[1] = (2048u & 0xFFFFu) << 16;                       // tensor_dim0[15:0]
  g1[2] = (2048u >> 16) | (1u << 16);                    // tensor_dim0[31:16] | tensor_dim1=1
  g1[3] = (2048u << 16);                                 // tile_dim0=2048
  g1[4] = 0u;                                            // tile_dim1/2 unused
  g1[5] = 2048u;                                         // tensor_dim0_stride[31:0]
  g1[6] = 0u;
  g1[7] = 0u;
  asm volatile("tensor_load_to_lds %0, %1" :: "s"(g0), "s"(g1) : "memory");
}

// ---------- prep: pack codebook as bf16 hi/lo + squared norms (runs once per call) ----------
__global__ __launch_bounds__(THREADS)
void vq_prep_kernel(const float* __restrict__ emb,
                    unsigned* __restrict__ ebf, float* __restrict__ esq) {
  const int code = blockIdx.x * THREADS + threadIdx.x;
  if (code >= N_EMB) return;
  const float* er = emb + (size_t)code * CDIM;
  unsigned* o = ebf + (size_t)code * CODE_GUINTS;
  float acc = 0.f;
  #pragma unroll
  for (int i = 0; i < 32; ++i) {
    float x0 = er[2 * i], x1 = er[2 * i + 1];
    acc += x0 * x0 + x1 * x1;
    unsigned h0 = f2bf(x0), h1 = f2bf(x1);
    o[i] = h0 | (h1 << 16);                                   // hi bf16, k=2i,2i+1
    unsigned l0 = f2bf(x0 - bf2f(h0)), l1 = f2bf(x1 - bf2f(h1));
    o[32 + i] = l0 | (l1 << 16);                              // lo bf16
  }
  esq[code] = acc;
}

// ---------- main ----------
__global__ __launch_bounds__(THREADS)
void vq_main_kernel(const float* __restrict__ z, const float* __restrict__ emb,
                    const unsigned* __restrict__ ebf, const float* __restrict__ esq_g,
                    float* __restrict__ out) {
  __shared__ __align__(16) unsigned char smem[SMEM_BYTES];
  float* lds_z   = (float*)smem;                      // phase 1 only; reused as chunk bufs
  float* lds_esq = (float*)(smem + SMEM_ESQ_OFF);
  int*   lds_idx = (int*)(smem + SMEM_IDX_OFF);

  const int tid = threadIdx.x;
  const int bid = blockIdx.x;
  const int b   = bid >> 5;                  // SEQ/128 = 32 blocks per batch
  const int s0  = (bid & 31) * ROWS_PER_BLOCK;
  const float* zb = z + (size_t)b * CDIM * SEQ + s0;

  // stage z tile transposed into LDS rows
  {
    const int sl = tid & 127, c0 = tid >> 7;
    #pragma unroll
    for (int c = c0; c < CDIM; c += 2)
      lds_z[sl * ZP + c] = zb[(size_t)c * SEQ + sl];
  }
  for (int t = tid; t < N_EMB; t += THREADS) lds_esq[t] = esq_g[t];
  __syncthreads();

  const int lane = tid & 31;
  const int w    = tid >> 5;
  const int m    = lane & 15;
  const int kh   = lane >> 4;

  // A fragments (16 z rows x K=64) as bf16 hi/lo, built once, held in VGPRs.
  // A 16-bit layout: lanes 0-15: V0-7=K0-15; lanes 16-31: K16-31 (contiguous, matches B).
  v16bf ah[2], al[2];
  {
    const float* zr = &lds_z[(w * 16 + m) * ZP];
    #pragma unroll
    for (int q = 0; q < 2; ++q)
      #pragma unroll
      for (int v = 0; v < 8; ++v)
        #pragma unroll
        for (int p = 0; p < 2; ++p) {
          int k = q * 32 + kh * 16 + v * 2 + p;
          float x = zr[k];
          unsigned hb = f2bf(x);
          ah[q][2 * v + p] = __builtin_bit_cast(__bf16, (unsigned short)hb);
          al[q][2 * v + p] = __builtin_bit_cast(__bf16, (unsigned short)f2bf(x - bf2f(hb)));
        }
  }
  __syncthreads();   // z region dead -> reuse as double-buffered codebook chunks

  const unsigned lds_base = (unsigned)(unsigned long long)(uintptr_t)smem;

  float best[8];
  int   bidx[8];
  #pragma unroll
  for (int v = 0; v < 8; ++v) { best[v] = 3.4e38f; bidx[v] = 0; }

  // prologue: TDM chunk 0 into buffer 0
  if (w == 0) {
    tdm_load_chunk(ebf, lds_base);
    __builtin_amdgcn_s_wait_tensorcnt(0);
  }
  __syncthreads();

  for (int cc = 0; cc < NCHUNK; ++cc) {
    const int cur = cc & 1;
    if (w == 0 && cc + 1 < NCHUNK)   // prefetch next chunk, overlapped with WMMAs
      tdm_load_chunk(ebf + (size_t)(cc + 1) * CHUNK_GUINTS,
                     lds_base + (unsigned)((cur ^ 1) * CHUNK_LDS_BYTES));

    const unsigned* buf = (const unsigned*)(smem + cur * CHUNK_LDS_BYTES);

    #pragma unroll
    for (int j = 0; j < CHUNK_CODES / 16; ++j) {
      // B fragments straight from prepacked bf16: no conversion VALU.
      const unsigned* er = buf + (j * 16 + m) * EPW + kh * 8;
      v16bf bh0 = ldbf16x16(er);        // K q=0 hi
      v16bf bh1 = ldbf16x16(er + 16);   // K q=1 hi
      v16bf bl0 = ldbf16x16(er + 32);   // K q=0 lo
      v16bf bl1 = ldbf16x16(er + 48);   // K q=1 lo

      v8f c = {0.f, 0.f, 0.f, 0.f, 0.f, 0.f, 0.f, 0.f};
      c = __builtin_amdgcn_wmma_f32_16x16x32_bf16(false, ah[0], false, bh0, (short)0, c, false, false);
      c = __builtin_amdgcn_wmma_f32_16x16x32_bf16(false, ah[1], false, bh1, (short)0, c, false, false);
      c = __builtin_amdgcn_wmma_f32_16x16x32_bf16(false, ah[0], false, bl0, (short)0, c, false, false);
      c = __builtin_amdgcn_wmma_f32_16x16x32_bf16(false, ah[1], false, bl1, (short)0, c, false, false);
      c = __builtin_amdgcn_wmma_f32_16x16x32_bf16(false, al[0], false, bh0, (short)0, c, false, false);
      c = __builtin_amdgcn_wmma_f32_16x16x32_bf16(false, al[1], false, bh1, (short)0, c, false, false);

      const int gcode = cc * CHUNK_CODES + j * 16 + m;
      const float es  = lds_esq[gcode];
      #pragma unroll
      for (int v = 0; v < 8; ++v) {
        float d = es - 2.0f * c[v];     // ||e||^2 - 2 z.e
        if (d < best[v]) { best[v] = d; bidx[v] = gcode; }
      }
    }

    if (w == 0) __builtin_amdgcn_s_wait_tensorcnt(0);
    __syncthreads();
  }

  // argmin across the 16 lanes of each half-wave (wave32-safe shfl widths)
  #pragma unroll
  for (int v = 0; v < 8; ++v) {
    float d  = best[v];
    int   ix = bidx[v];
    #pragma unroll
    for (int off = 1; off < 16; off <<= 1) {
      float od  = __shfl_xor(d, off, 32);
      int   oix = __shfl_xor(ix, off, 32);
      if (od < d || (od == d && oix < ix)) { d = od; ix = oix; }
    }
    if (m == 0) lds_idx[w * 16 + kh * 8 + v] = ix;
  }
  __syncthreads();

  // gather emb[idx] rows and write output transposed back to (b, c, s)
  {
    const int sl = tid & 127, c0 = tid >> 7;
    const int idx = lds_idx[sl];
    const float* erow = emb + (size_t)idx * CDIM;
    float* ob = out + (size_t)b * CDIM * SEQ + s0 + sl;
    #pragma unroll
    for (int c = c0; c < CDIM; c += 2)
      ob[(size_t)c * SEQ] = erow[c];
  }
}

extern "C" void kernel_launch(void* const* d_in, const int* in_sizes, int n_in,
                              void* d_out, int out_size, void* d_ws, size_t ws_size,
                              hipStream_t stream) {
  const float* z   = (const float*)d_in[0];   // (32, 64, 4096) f32
  const float* emb = (const float*)d_in[1];   // (1024, 64) f32
  float* out = (float*)d_out;                 // (32, 64, 4096) f32
  (void)in_sizes; (void)n_in; (void)out_size; (void)ws_size;

  unsigned* ebf = (unsigned*)d_ws;                          // 256 KiB packed bf16 hi/lo
  float*    esq = (float*)((char*)d_ws + N_EMB * CODE_GUINTS * 4);  // +4 KiB norms

  vq_prep_kernel<<<dim3(N_EMB / THREADS), dim3(THREADS), 0, stream>>>(emb, ebf, esq);
  vq_main_kernel<<<dim3(BS * SEQ / ROWS_PER_BLOCK), dim3(THREADS), 0, stream>>>(
      z, emb, ebf, esq, out);
}
